// TopKSAE_59785944760741
// MI455X (gfx1250) — matrix operand order repaired and checked by reading, not compiled, not run on hardware
//
#include <hip/hip_runtime.h>

#define N_ROWS 16384
#define D_IN   768
#define D_LAT  12288
#define TOPK   32

typedef __attribute__((ext_vector_type(16))) __bf16 v16bf;
typedef __attribute__((ext_vector_type(8)))  float  v8f;

union FragAB { v16bf v; int4 q[2]; };

// ---------- gfx1250 async global->LDS copy (ASYNCcnt path) ----------
#ifndef USE_ASYNC_LDS
#define USE_ASYNC_LDS 1
#endif

#if USE_ASYNC_LDS && __has_builtin(__builtin_amdgcn_global_load_async_to_lds_b128)
#define HAVE_ASYNC 1
#else
#define HAVE_ASYNC 0
#endif

// native clang vector type matching the builtin prototype:
//   '__attribute__((__vector_size__(4 * sizeof(int)))) int'
typedef int nv4i __attribute__((vector_size(16)));
typedef __attribute__((address_space(1))) nv4i gas_v4i;
typedef __attribute__((address_space(3))) nv4i las_v4i;

__device__ __forceinline__ void copy16(char* l, const __bf16* g) {
#if HAVE_ASYNC
  __builtin_amdgcn_global_load_async_to_lds_b128(
      (gas_v4i*)(void*)g, (las_v4i*)(void*)l, /*offset=*/0, /*cpol=*/0);
#else
  *(int4*)l = *(const int4*)g;
#endif
}

__device__ __forceinline__ void waitAsync() {
#if HAVE_ASYNC
#if __has_builtin(__builtin_amdgcn_s_wait_asynccnt)
  __builtin_amdgcn_s_wait_asynccnt(0);
#else
  asm volatile("s_wait_asynccnt 0x0" ::: "memory");
#endif
#endif
}

// ---------------- kernel 1: f32 -> bf16 (vectorized x4) ----------------
__global__ void cvt_bf16_kernel(const float* __restrict__ src,
                                __bf16* __restrict__ dst, int n4) {
  int i = blockIdx.x * blockDim.x + threadIdx.x;
  if (i >= n4) return;
  float4 f = ((const float4*)src)[i];
  union { __bf16 h[4]; uint2 u; } p;
  p.h[0] = (__bf16)f.x; p.h[1] = (__bf16)f.y;
  p.h[2] = (__bf16)f.z; p.h[3] = (__bf16)f.w;
  *((uint2*)(dst + (size_t)i * 4)) = p.u;
}

// ---------------- kernel 2: W_dec [D_IN][D_LAT] -> wdT [D_LAT][D_IN] ----
__global__ void transpose_kernel(const float* __restrict__ in,
                                 float* __restrict__ out) {
  __shared__ float tile[32][33];
  const int l  = blockIdx.x * 32 + threadIdx.x;
  const int d0 = blockIdx.y * 32;
  for (int jj = 0; jj < 4; ++jj) {
    int j = threadIdx.y + jj * 8;
    tile[j][threadIdx.x] = in[(size_t)(d0 + j) * D_LAT + l];
  }
  __syncthreads();
  const int d  = d0 + threadIdx.x;
  const int l0 = blockIdx.x * 32;
  for (int jj = 0; jj < 4; ++jj) {
    int j = threadIdx.y + jj * 8;
    out[(size_t)(l0 + j) * D_IN + d] = tile[threadIdx.x][j];
  }
}

// ---------------- kernel 3: encode GEMM, z = relu(x @ W_enc^T) ----------
// 256x128 macro tile / WG (8 waves). Wave owns 32 rows (2 A-frags) x 128
// cols (8 B-frags) -> 16 WMMA per 32-wide k-step, 24 k-steps.
// Double-buffered LDS, one barrier per k-step; fills go through the
// gfx1250 async-to-LDS path when available.
#define LDSA_BYTES 20480
#define LDSB_BYTES 10240
#define BUF_BYTES  (LDSA_BYTES + LDSB_BYTES)

__device__ __forceinline__ void fill_tiles(char* ldsA, char* ldsB,
                                           const __bf16* __restrict__ xh,
                                           const __bf16* __restrict__ wh,
                                           int mBase, int nBase, int kb,
                                           int tid) {
  // A: 256 rows * 4 chunks of 16B = 1024 chunks
#pragma unroll
  for (int u = 0; u < 4; ++u) {
    int c = tid + u * 256;
    int row = c >> 2, co = c & 3;
    copy16(ldsA + row * 80 + co * 16,
           xh + (size_t)(mBase + row) * D_IN + kb + co * 8);
  }
  // B: 128 rows * 4 chunks = 512 chunks
#pragma unroll
  for (int u = 0; u < 2; ++u) {
    int c = tid + u * 256;
    int row = c >> 2, co = c & 3;
    copy16(ldsB + row * 80 + co * 16,
           wh + (size_t)(nBase + row) * D_IN + kb + co * 8);
  }
}

__global__ __launch_bounds__(256) void encode_wmma_kernel(
    const __bf16* __restrict__ xh, const __bf16* __restrict__ wh,
    float* __restrict__ zout) {
  __shared__ int4 smemq[2 * BUF_BYTES / 16];
  char* smem = (char*)smemq;

  const int tid   = threadIdx.x;
  const int lane  = tid & 31;
  const int wave  = tid >> 5;      // 0..7
  const int lr    = lane & 15;
  const int hi    = lane >> 4;     // half-wave select
  const int mBase = blockIdx.x * 256;
  const int nBase = blockIdx.y * 128;

  v8f acc[2][8];
  const v8f vz = {0.f, 0.f, 0.f, 0.f, 0.f, 0.f, 0.f, 0.f};
#pragma unroll
  for (int s = 0; s < 2; ++s)
#pragma unroll
    for (int c = 0; c < 8; ++c) acc[s][c] = vz;

  const int NSTEP = D_IN / 32;  // 24

  // prologue: fill buffer 0
  fill_tiles(smem, smem + LDSA_BYTES, xh, wh, mBase, nBase, 0, tid);
  waitAsync();
  __syncthreads();

  for (int step = 0; step < NSTEP; ++step) {
    const int cur = step & 1;
    char* curA = smem + cur * BUF_BYTES;
    char* curB = curA + LDSA_BYTES;
    if (step + 1 < NSTEP) {
      char* nxtA = smem + (1 - cur) * BUF_BYTES;
      fill_tiles(nxtA, nxtA + LDSA_BYTES, xh, wh, mBase, nBase,
                 (step + 1) * 32, tid);
    }

    // A fragments for the wave's two 16-row subtiles
    FragAB af[2];
#pragma unroll
    for (int s = 0; s < 2; ++s) {
      const char* ap = curA + (wave * 32 + s * 16 + lr) * 80 + hi * 16;
      af[s].q[0] = *(const int4*)(ap);
      af[s].q[1] = *(const int4*)(ap + 32);
    }
#pragma unroll
    for (int c = 0; c < 8; ++c) {
      FragAB bf;
      const char* bp = curB + (c * 16 + lr) * 80 + hi * 16;
      bf.q[0] = *(const int4*)(bp);
      bf.q[1] = *(const int4*)(bp + 32);
#pragma unroll
      for (int s = 0; s < 2; ++s)
        acc[s][c] = __builtin_amdgcn_wmma_f32_16x16x32_bf16(
            false, af[s].v, false, bf.v, (short)0, acc[s][c], false, false);
    }

    waitAsync();       // next-buffer async fills complete
    __syncthreads();   // all waves done reading cur before it is refilled
  }

  // epilogue: ReLU + store (C/D layout: vgpr j -> M=j+hi*8, lane%16 -> N)
#pragma unroll
  for (int s = 0; s < 2; ++s) {
#pragma unroll
    for (int c = 0; c < 8; ++c) {
#pragma unroll
      for (int j = 0; j < 8; ++j) {
        int m = mBase + wave * 32 + s * 16 + hi * 8 + j;
        int n = nBase + c * 16 + lr;
        float v = acc[s][c][j];
        zout[(size_t)m * D_LAT + n] = v > 0.f ? v : 0.f;
      }
    }
  }
}

// ---------------- kernel 4: per-row exact radix top-K + mask + compact ---
__global__ __launch_bounds__(256) void topk_kernel(float* __restrict__ z,
                                                   float* __restrict__ vals,
                                                   int* __restrict__ idxs) {
  __shared__ float row[D_LAT];          // 48 KB
  __shared__ unsigned hist[256];
  __shared__ int cntGt[256], cntEq[256];
  __shared__ int baseGt[257], baseEq[257];
  __shared__ unsigned sPrefix;
  __shared__ int sKr;

  const int tid = threadIdx.x;
  const size_t rbase = (size_t)blockIdx.x * D_LAT;

  for (int i = tid; i < D_LAT; i += 256) row[i] = z[rbase + i];
  __syncthreads();

  unsigned prefix = 0;
  int Kr = TOPK;
  for (int pass = 3; pass >= 0; --pass) {
    const int shift = pass * 8;
    hist[tid] = 0;
    __syncthreads();
    for (int i = tid; i < D_LAT; i += 256) {
      unsigned u = __float_as_uint(row[i]);
      bool cand = (pass == 3) || ((u >> (shift + 8)) == prefix);
      if (cand) atomicAdd(&hist[(u >> shift) & 0xFF], 1u);
    }
    __syncthreads();
    if (tid == 0) {
      int d = 255;
      unsigned run = 0;
      for (; d >= 0; --d) {
        run += hist[d];
        if ((int)run >= Kr) break;
      }
      if (d < 0) d = 0;
      sKr = Kr - (int)(run - hist[d]);
      sPrefix = (prefix << 8) | (unsigned)d;
    }
    __syncthreads();
    prefix = sPrefix;
    Kr = sKr;
    __syncthreads();
  }
  const unsigned T = prefix;

  const int per = D_LAT / 256;  // 48
  const int begin = tid * per, end = begin + per;
  int lgt = 0, leq = 0;
  for (int i = begin; i < end; ++i) {
    unsigned u = __float_as_uint(row[i]);
    if (u > T) lgt++;
    else if (u == T) leq++;
  }
  cntGt[tid] = lgt;
  cntEq[tid] = leq;
  __syncthreads();
  if (tid == 0) {
    int g = 0, e = 0;
    for (int t = 0; t < 256; ++t) {
      baseGt[t] = g; baseEq[t] = e;
      g += cntGt[t]; e += cntEq[t];
    }
    baseGt[256] = g; baseEq[256] = e;
  }
  __syncthreads();
  const int needTies = TOPK - baseGt[256];
  int g = baseGt[tid], e = baseEq[tid];
  for (int i = begin; i < end; ++i) {
    float v = row[i];
    unsigned u = __float_as_uint(v);
    bool keep = false;
    int pos = 0;
    if (u > T) {
      pos = g + (e < needTies ? e : needTies);
      keep = true;
      g++;
    } else if (u == T) {
      if (e < needTies) { pos = g + e; keep = true; }
      e++;
    }
    if (keep) {
      vals[(size_t)blockIdx.x * TOPK + pos] = v;
      idxs[(size_t)blockIdx.x * TOPK + pos] = i;
    }
    z[rbase + i] = keep ? v : 0.f;
  }
}

// ---------------- kernel 5: sparse decode, x_hat = z_sparse @ W_dec^T ----
__global__ __launch_bounds__(256) void decode_kernel(
    const float* __restrict__ wdT, const float* __restrict__ vals,
    const int* __restrict__ idxs, float* __restrict__ xhat) {
  __shared__ float sv[TOPK];
  __shared__ int si[TOPK];
  const int tid = threadIdx.x;
  const int r = blockIdx.x;
  if (tid < TOPK) {
    sv[tid] = vals[(size_t)r * TOPK + tid];
    si[tid] = idxs[(size_t)r * TOPK + tid];
  }
  __syncthreads();
  for (int d = tid; d < D_IN; d += 256) {
    float acc = 0.f;
#pragma unroll
    for (int k = 0; k < TOPK; ++k)
      acc = fmaf(sv[k], wdT[(size_t)si[k] * D_IN + d], acc);
    xhat[(size_t)r * D_IN + d] = acc;
  }
}

extern "C" void kernel_launch(void* const* d_in, const int* in_sizes, int n_in,
                              void* d_out, int out_size, void* d_ws,
                              size_t ws_size, hipStream_t stream) {
  const float* x    = (const float*)d_in[0];
  const float* Wenc = (const float*)d_in[1];
  const float* Wdec = (const float*)d_in[2];

  float* xhat = (float*)d_out;                           // [N, D_IN]
  float* zout = (float*)d_out + (size_t)N_ROWS * D_IN;   // [N, D_LAT]

  char* ws = (char*)d_ws;
  const size_t xh_bytes  = (size_t)N_ROWS * D_IN * 2;
  const size_t wh_bytes  = (size_t)D_LAT * D_IN * 2;
  const size_t wdT_bytes = (size_t)D_LAT * D_IN * 4;
  const size_t val_bytes = (size_t)N_ROWS * TOPK * 4;
  __bf16* xh  = (__bf16*)(ws);
  __bf16* wh  = (__bf16*)(ws + xh_bytes);
  float*  wdT = (float*)(ws + xh_bytes + wh_bytes);
  float*  vals = (float*)(ws + xh_bytes + wh_bytes + wdT_bytes);
  int*    idxs = (int*)(ws + xh_bytes + wh_bytes + wdT_bytes + val_bytes);

  {
    int n4 = N_ROWS * D_IN / 4;
    cvt_bf16_kernel<<<(n4 + 255) / 256, 256, 0, stream>>>(x, xh, n4);
  }
  {
    int n4 = D_LAT * D_IN / 4;
    cvt_bf16_kernel<<<(n4 + 255) / 256, 256, 0, stream>>>(Wenc, wh, n4);
  }
  transpose_kernel<<<dim3(D_LAT / 32, D_IN / 32), dim3(32, 8), 0, stream>>>(
      Wdec, wdT);
  encode_wmma_kernel<<<dim3(N_ROWS / 256, D_LAT / 128), 256, 0, stream>>>(
      xh, wh, zout);
  topk_kernel<<<N_ROWS, 256, 0, stream>>>(zout, vals, idxs);
  decode_kernel<<<N_ROWS, 256, 0, stream>>>(wdT, vals, idxs, xhat);
}
